// EGNNetRLBO_75806172774700
// MI455X (gfx1250) — matrix operand order repaired and checked by reading, not compiled, not run on hardware
//
#include <hip/hip_runtime.h>
#include <hip/hip_bf16.h>

typedef __attribute__((ext_vector_type(16))) _Float16 v16h;
typedef __attribute__((ext_vector_type(8)))  _Float16 v8h;
typedef __attribute__((ext_vector_type(8)))  float    v8f;

#define ASTRIDE 136   // 128 + 8 pad (halfs)  -> 272B rows, 16B aligned
#define MSTRIDE 72    // 64 + 8 pad  (halfs)  -> 144B rows, 16B aligned

__device__ __forceinline__ float silu_f(float v) {
    return v / (1.0f + __expf(-v));
}

__device__ __forceinline__ v8f wmma_f16(v16h a, v16h b, v8f c) {
    return __builtin_amdgcn_wmma_f32_16x16x32_f16(false, a, false, b, (short)0, c, false, false);
}

// concat two contiguous 8-half loads into one 16-half fragment
__device__ __forceinline__ v16h frag_cat(const _Float16* p0, const _Float16* p1) {
    v8h lo = *(const v8h*)p0;
    v8h hi = *(const v8h*)p1;
    v16h r;
#pragma unroll
    for (int i = 0; i < 8; ++i) { r[i] = lo[i]; r[i + 8] = hi[i]; }
    return r;
}

// A fragment (16xK tile in LDS, row-major, rows = M): 16x32 slab at kbase.
// ISA layout: lane m=l&15, hi=l>>4; elems 0..7 -> k=kbase+hi*8+j ; elems 8..15 -> k=kbase+16+hi*8+j
__device__ __forceinline__ v16h a_frag(const _Float16* tile, int stride, int kbase, int lane) {
    int m = lane & 15, hb = lane >> 4;
    const _Float16* p = tile + m * stride + kbase + hb * 8;
    return frag_cat(p, p + 16);
}

// B fragment from transposed weights Wt[n][k]: lane col n=l&15, elems 0..7 -> k = kb + (l>>4)*8 + e ; elems 8..15 -> k = kb + 16 + (l>>4)*8 + (e-8)
__device__ __forceinline__ v16h b_frag(const _Float16* wt, int stride, int nb, int kb, int lane) {
    int n = lane & 15, hb = lane >> 4;
    const _Float16* p = wt + (nb + n) * stride + kb + hb * 8;
    return frag_cat(p, p + 16);
}

// ---------------- feature projection: h = x @ W(14x64) + b ; also f16 shadow ----------------
__global__ __launch_bounds__(256) void proj_kernel(const float* __restrict__ x,
                                                   const float* __restrict__ W,
                                                   const float* __restrict__ b,
                                                   float* __restrict__ h,
                                                   _Float16* __restrict__ hh, int N_) {
    __shared__ float Ws[14 * 64];
    __shared__ float bs[64];
    for (int t = threadIdx.x; t < 14 * 64; t += 256) Ws[t] = W[t];
    if (threadIdx.x < 64) bs[threadIdx.x] = b[threadIdx.x];
    __syncthreads();
    int idx = blockIdx.x * 256 + threadIdx.x;
    if (idx >= N_ * 64) return;
    int node = idx >> 6, d = idx & 63;
    float s = bs[d];
#pragma unroll
    for (int k = 0; k < 14; ++k) s += x[node * 14 + k] * Ws[k * 64 + d];
    h[idx] = s;
    hh[idx] = (_Float16)s;
}

// ---------------- edge message + scatter-add: per block 128 edges, per wave 16 ----------------
__global__ __launch_bounds__(256) void edge_kernel(const _Float16* __restrict__ hh,
                                                   const int* __restrict__ ei,
                                                   const float* __restrict__ W1,
                                                   const float* __restrict__ B1,
                                                   const float* __restrict__ W2,
                                                   const float* __restrict__ B2,
                                                   float* __restrict__ agg, int E_) {
    __shared__ _Float16 Wt1[64 * ASTRIDE];        // lin1 transposed [n][k], K=128
    __shared__ _Float16 Wt2[64 * MSTRIDE];        // lin2 transposed [n][k], K=64
    __shared__ _Float16 smA[8 * 16 * ASTRIDE];    // per-wave A tiles (reused for m1)
    __shared__ float b1s[64], b2s[64];
    __shared__ int srcS[128], dstS[128];

    const int tid = threadIdx.x, lane = tid & 31, wave = tid >> 5;
    const int ebase = blockIdx.x * 128;

    for (int t = tid; t < 128 * 64; t += 256) { int k = t >> 6, n = t & 63; Wt1[n * ASTRIDE + k] = (_Float16)W1[t]; }
    for (int t = tid; t < 64 * 64; t += 256)  { int k = t >> 6, n = t & 63; Wt2[n * MSTRIDE + k] = (_Float16)W2[t]; }
    if (tid < 64) { b1s[tid] = B1[tid]; b2s[tid] = B2[tid]; }
    for (int t = tid; t < 128; t += 256) {
        int e = ebase + t;
        srcS[t] = (e < E_) ? ei[e] : 0;
        dstS[t] = (e < E_) ? ei[E_ + e] : 0;
    }
    __syncthreads();

    _Float16* As = smA + wave * (16 * ASTRIDE);
    // gather [h[dst] | h[src]]: 16 edges x 128 halfs = 256 chunks of 8 halfs / 32 lanes
#pragma unroll
    for (int it = 0; it < 8; ++it) {
        int c = it * 32 + lane;
        int el = c >> 4, part = c & 15;
        int eg = ebase + wave * 16 + el;
        int node = (part < 8) ? dstS[wave * 16 + el] : srcS[wave * 16 + el];
        v8h v = {};
        if (eg < E_) v = *(const v8h*)(hh + (long)node * 64 + (part & 7) * 8);
        *(v8h*)(As + el * ASTRIDE + part * 8) = v;
    }
    __syncthreads();

    v16h a0 = a_frag(As, ASTRIDE, 0, lane);
    v16h a1 = a_frag(As, ASTRIDE, 32, lane);
    v16h a2 = a_frag(As, ASTRIDE, 64, lane);
    v16h a3 = a_frag(As, ASTRIDE, 96, lane);

    v8f acc[4];
#pragma unroll
    for (int nt = 0; nt < 4; ++nt) {
        v8f c = {};
        c = wmma_f16(a0, b_frag(Wt1, ASTRIDE, nt * 16, 0, lane), c);
        c = wmma_f16(a1, b_frag(Wt1, ASTRIDE, nt * 16, 32, lane), c);
        c = wmma_f16(a2, b_frag(Wt1, ASTRIDE, nt * 16, 64, lane), c);
        c = wmma_f16(a3, b_frag(Wt1, ASTRIDE, nt * 16, 96, lane), c);
        acc[nt] = c;
    }
    __syncthreads();

    // silu(m1) -> LDS (reuse As region) in A layout (row-major 16x64)
    _Float16* Ms = smA + wave * (16 * ASTRIDE);
    int ln = lane & 15, hb = lane >> 4;
#pragma unroll
    for (int nt = 0; nt < 4; ++nt) {
        float bias = b1s[nt * 16 + ln];
#pragma unroll
        for (int r = 0; r < 8; ++r) {
            float v = silu_f(acc[nt][r] + bias);
            Ms[(r + 8 * hb) * MSTRIDE + nt * 16 + ln] = (_Float16)v;
        }
    }
    __syncthreads();

    v16h q0 = a_frag(Ms, MSTRIDE, 0, lane);
    v16h q1 = a_frag(Ms, MSTRIDE, 32, lane);
#pragma unroll
    for (int nt = 0; nt < 4; ++nt) {
        v8f c = {};
        c = wmma_f16(q0, b_frag(Wt2, MSTRIDE, nt * 16, 0, lane), c);
        c = wmma_f16(q1, b_frag(Wt2, MSTRIDE, nt * 16, 32, lane), c);
        float bias = b2s[nt * 16 + ln];
#pragma unroll
        for (int r = 0; r < 8; ++r) {
            int m = r + 8 * hb;
            int eg = ebase + wave * 16 + m;
            if (eg < E_) {
                float v = silu_f(c[r] + bias);
                atomicAdd(&agg[(long)dstS[wave * 16 + m] * 64 + nt * 16 + ln], v);
            }
        }
    }
}

// ---------------- node update: h = node_mlp([h|agg]) + h ; per block 128 nodes ----------------
__global__ __launch_bounds__(256) void node_kernel(float* __restrict__ h,
                                                   _Float16* __restrict__ hh,
                                                   const float* __restrict__ agg,
                                                   const float* __restrict__ W1,
                                                   const float* __restrict__ B1,
                                                   const float* __restrict__ W2,
                                                   const float* __restrict__ B2,
                                                   int N_) {
    __shared__ _Float16 Wt1[64 * ASTRIDE];
    __shared__ _Float16 Wt2[64 * MSTRIDE];
    __shared__ _Float16 smA[8 * 16 * ASTRIDE];
    __shared__ float b1s[64], b2s[64];

    const int tid = threadIdx.x, lane = tid & 31, wave = tid >> 5;
    const int rbase = blockIdx.x * 128 + wave * 16;

    for (int t = tid; t < 128 * 64; t += 256) { int k = t >> 6, n = t & 63; Wt1[n * ASTRIDE + k] = (_Float16)W1[t]; }
    for (int t = tid; t < 64 * 64; t += 256)  { int k = t >> 6, n = t & 63; Wt2[n * MSTRIDE + k] = (_Float16)W2[t]; }
    if (tid < 64) { b1s[tid] = B1[tid]; b2s[tid] = B2[tid]; }
    __syncthreads();

    _Float16* As = smA + wave * (16 * ASTRIDE);
#pragma unroll
    for (int it = 0; it < 8; ++it) {
        int c = it * 32 + lane;
        int rl = c >> 4, part = c & 15;
        int row = rbase + rl;
        v8h v = {};
        if (row < N_) {
            if (part < 8) {
                v = *(const v8h*)(hh + (long)row * 64 + part * 8);
            } else {
                const float* p = agg + (long)row * 64 + (part - 8) * 8;
#pragma unroll
                for (int i = 0; i < 8; ++i) v[i] = (_Float16)p[i];
            }
        }
        *(v8h*)(As + rl * ASTRIDE + part * 8) = v;
    }
    __syncthreads();

    v16h a0 = a_frag(As, ASTRIDE, 0, lane);
    v16h a1 = a_frag(As, ASTRIDE, 32, lane);
    v16h a2 = a_frag(As, ASTRIDE, 64, lane);
    v16h a3 = a_frag(As, ASTRIDE, 96, lane);

    v8f acc[4];
#pragma unroll
    for (int nt = 0; nt < 4; ++nt) {
        v8f c = {};
        c = wmma_f16(a0, b_frag(Wt1, ASTRIDE, nt * 16, 0, lane), c);
        c = wmma_f16(a1, b_frag(Wt1, ASTRIDE, nt * 16, 32, lane), c);
        c = wmma_f16(a2, b_frag(Wt1, ASTRIDE, nt * 16, 64, lane), c);
        c = wmma_f16(a3, b_frag(Wt1, ASTRIDE, nt * 16, 96, lane), c);
        acc[nt] = c;
    }
    __syncthreads();

    _Float16* Ms = smA + wave * (16 * ASTRIDE);
    int ln = lane & 15, hb = lane >> 4;
#pragma unroll
    for (int nt = 0; nt < 4; ++nt) {
        float bias = b1s[nt * 16 + ln];
#pragma unroll
        for (int r = 0; r < 8; ++r) {
            float v = silu_f(acc[nt][r] + bias);
            Ms[(r + 8 * hb) * MSTRIDE + nt * 16 + ln] = (_Float16)v;
        }
    }
    __syncthreads();

    v16h q0 = a_frag(Ms, MSTRIDE, 0, lane);
    v16h q1 = a_frag(Ms, MSTRIDE, 32, lane);
#pragma unroll
    for (int nt = 0; nt < 4; ++nt) {
        v8f c = {};
        c = wmma_f16(q0, b_frag(Wt2, MSTRIDE, nt * 16, 0, lane), c);
        c = wmma_f16(q1, b_frag(Wt2, MSTRIDE, nt * 16, 32, lane), c);
        float bias = b2s[nt * 16 + ln];
#pragma unroll
        for (int r = 0; r < 8; ++r) {
            int row = rbase + r + 8 * hb;
            if (row < N_) {
                long o = (long)row * 64 + nt * 16 + ln;
                float v = c[r] + bias + h[o];   // residual
                h[o] = v;
                hh[o] = (_Float16)v;
            }
        }
    }
}

// ---------------- readout stage 1: o = silu(h@lin1+b)@lin2+b, scatter-add per graph ----------------
__global__ __launch_bounds__(256) void readout1_kernel(const _Float16* __restrict__ hh,
                                                       const int* __restrict__ batch,
                                                       const float* __restrict__ W1,
                                                       const float* __restrict__ B1,
                                                       const float* __restrict__ W2,
                                                       const float* __restrict__ B2,
                                                       float* __restrict__ og, int N_) {
    __shared__ _Float16 Wt1[64 * MSTRIDE];
    __shared__ _Float16 Wt2[64 * MSTRIDE];
    __shared__ _Float16 smA[8 * 16 * MSTRIDE];
    __shared__ float b1s[64], b2s[64];
    __shared__ int batS[128];

    const int tid = threadIdx.x, lane = tid & 31, wave = tid >> 5;
    const int rbase = blockIdx.x * 128 + wave * 16;

    for (int t = tid; t < 64 * 64; t += 256) {
        int k = t >> 6, n = t & 63;
        Wt1[n * MSTRIDE + k] = (_Float16)W1[t];
        Wt2[n * MSTRIDE + k] = (_Float16)W2[t];
    }
    if (tid < 64) { b1s[tid] = B1[tid]; b2s[tid] = B2[tid]; }
    for (int t = tid; t < 128; t += 256) {
        int row = blockIdx.x * 128 + t;
        batS[t] = (row < N_) ? batch[row] : 0;
    }
    __syncthreads();

    _Float16* As = smA + wave * (16 * MSTRIDE);
#pragma unroll
    for (int it = 0; it < 4; ++it) {
        int c = it * 32 + lane;
        int rl = c >> 3, part = c & 7;    // 8 chunks per row (64 halfs)
        int row = rbase + rl;
        v8h v = {};
        if (row < N_) v = *(const v8h*)(hh + (long)row * 64 + part * 8);
        *(v8h*)(As + rl * MSTRIDE + part * 8) = v;
    }
    __syncthreads();

    v16h a0 = a_frag(As, MSTRIDE, 0, lane);
    v16h a1 = a_frag(As, MSTRIDE, 32, lane);

    v8f acc[4];
#pragma unroll
    for (int nt = 0; nt < 4; ++nt) {
        v8f c = {};
        c = wmma_f16(a0, b_frag(Wt1, MSTRIDE, nt * 16, 0, lane), c);
        c = wmma_f16(a1, b_frag(Wt1, MSTRIDE, nt * 16, 32, lane), c);
        acc[nt] = c;
    }
    __syncthreads();

    _Float16* Ms = smA + wave * (16 * MSTRIDE);
    int ln = lane & 15, hb = lane >> 4;
#pragma unroll
    for (int nt = 0; nt < 4; ++nt) {
        float bias = b1s[nt * 16 + ln];
#pragma unroll
        for (int r = 0; r < 8; ++r) {
            float v = silu_f(acc[nt][r] + bias);
            Ms[(r + 8 * hb) * MSTRIDE + nt * 16 + ln] = (_Float16)v;
        }
    }
    __syncthreads();

    v16h q0 = a_frag(Ms, MSTRIDE, 0, lane);
    v16h q1 = a_frag(Ms, MSTRIDE, 32, lane);
#pragma unroll
    for (int nt = 0; nt < 4; ++nt) {
        v8f c = {};
        c = wmma_f16(q0, b_frag(Wt2, MSTRIDE, nt * 16, 0, lane), c);
        c = wmma_f16(q1, b_frag(Wt2, MSTRIDE, nt * 16, 32, lane), c);
        float bias = b2s[nt * 16 + ln];
#pragma unroll
        for (int r = 0; r < 8; ++r) {
            int m = r + 8 * hb;
            int row = rbase + m;
            if (row < N_) {
                float v = c[r] + bias;   // no silu on second layer
                atomicAdd(&og[(long)batS[wave * 16 + m] * 64 + nt * 16 + ln], v);
            }
        }
    }
}

// ---------------- readout stage 2: out[g] = silu(og@lin3+b3) @ lin4 + b4 ----------------
__global__ __launch_bounds__(64) void readout2_kernel(const float* __restrict__ og,
                                                      const float* __restrict__ W3,
                                                      const float* __restrict__ B3,
                                                      const float* __restrict__ W4,
                                                      const float* __restrict__ B4,
                                                      float* __restrict__ out) {
    __shared__ float gs[64];
    __shared__ float red[64];
    int g = blockIdx.x, t = threadIdx.x;
    gs[t] = og[(long)g * 64 + t];
    __syncthreads();
    float s = 0.f;
#pragma unroll 8
    for (int k = 0; k < 64; ++k) s += gs[k] * W3[k * 64 + t];
    s = silu_f(s + B3[t]);
    red[t] = s * W4[t];
    __syncthreads();
    for (int off = 32; off > 0; off >>= 1) {
        if (t < off) red[t] += red[t + off];
        __syncthreads();
    }
    if (t == 0) out[g] = red[0] + B4[0];
}

__global__ __launch_bounds__(256) void zero_kernel(float* __restrict__ p, int n) {
    int i = blockIdx.x * 256 + threadIdx.x;
    if (i < n) p[i] = 0.f;
}

__global__ __launch_bounds__(256) void copy_kernel(const float* __restrict__ s,
                                                   float* __restrict__ d, int n) {
    int i = blockIdx.x * 256 + threadIdx.x;
    if (i < n) d[i] = s[i];
}

extern "C" void kernel_launch(void* const* d_in, const int* in_sizes, int n_in,
                              void* d_out, int out_size, void* d_ws, size_t ws_size,
                              hipStream_t stream) {
    const float* x       = (const float*)d_in[0];
    // d_in[1] = pos (unused by reference)
    const int*   ei      = (const int*)d_in[2];
    const int*   batch   = (const int*)d_in[3];
    const float* proj_w  = (const float*)d_in[4];
    const float* proj_b  = (const float*)d_in[5];
    const float* edge_w1 = (const float*)d_in[6];
    const float* edge_b1 = (const float*)d_in[7];
    const float* edge_w2 = (const float*)d_in[8];
    const float* edge_b2 = (const float*)d_in[9];
    const float* node_w1 = (const float*)d_in[10];
    const float* node_b1 = (const float*)d_in[11];
    const float* node_w2 = (const float*)d_in[12];
    const float* node_b2 = (const float*)d_in[13];
    const float* lin1_w  = (const float*)d_in[14];
    const float* lin1_b  = (const float*)d_in[15];
    const float* lin2_w  = (const float*)d_in[16];
    const float* lin2_b  = (const float*)d_in[17];
    const float* lin3_w  = (const float*)d_in[18];
    const float* lin3_b  = (const float*)d_in[19];
    const float* lin4_w  = (const float*)d_in[20];
    const float* lin4_b  = (const float*)d_in[21];

    const int N_ = in_sizes[0] / 14;
    const int E_ = in_sizes[2] / 2;
    const int G_ = out_size - N_ * 64;
    const int L_ = in_sizes[6] / (128 * 64);

    float*    h   = (float*)d_ws;
    float*    agg = h + (long)N_ * 64;
    float*    og  = agg + (long)N_ * 64;
    _Float16* hh  = (_Float16*)(og + (long)G_ * 64);

    float* out = (float*)d_out;

    const int nd      = N_ * 64;
    const int ndBlk   = (nd + 255) / 256;
    const int eBlocks = (E_ + 127) / 128;
    const int nBlocks = (N_ + 127) / 128;

    proj_kernel<<<ndBlk, 256, 0, stream>>>(x, proj_w, proj_b, h, hh, N_);

    for (int l = 0; l < L_; ++l) {
        zero_kernel<<<ndBlk, 256, 0, stream>>>(agg, nd);
        edge_kernel<<<eBlocks, 256, 0, stream>>>(hh, ei,
                                                 edge_w1 + (long)l * 128 * 64, edge_b1 + l * 64,
                                                 edge_w2 + (long)l * 64 * 64,  edge_b2 + l * 64,
                                                 agg, E_);
        node_kernel<<<nBlocks, 256, 0, stream>>>(h, hh, agg,
                                                 node_w1 + (long)l * 128 * 64, node_b1 + l * 64,
                                                 node_w2 + (long)l * 64 * 64,  node_b2 + l * 64,
                                                 N_);
    }

    zero_kernel<<<(G_ * 64 + 255) / 256, 256, 0, stream>>>(og, G_ * 64);
    readout1_kernel<<<nBlocks, 256, 0, stream>>>(hh, batch, lin1_w, lin1_b, lin2_w, lin2_b, og, N_);
    readout2_kernel<<<G_, 64, 0, stream>>>(og, lin3_w, lin3_b, lin4_w, lin4_b, out);
    copy_kernel<<<ndBlk, 256, 0, stream>>>(h, out + G_, nd);
}
